// MHA_15015205667368
// MI455X (gfx1250) — compile-verified, matrix operand order checked by reading, not run on hardware
//
#include <hip/hip_runtime.h>
#include <hip/hip_bf16.h>

// ===========================================================================
// MHA forward on MI455X (gfx1250): bf16 WMMA everywhere, TDM tile staging.
//
// Pipeline (all matmul FLOPs through v_wmma_f32_16x16x32_bf16):
//   pack x/Wqkv/Wout -> bf16 (W pre-paired along K)        [one pass, VALU]
//   qkv  = x @ Wqkv + b      (TDM double-buffered GEMM, bf16 out)
//   RoPE on q,k (packed bf16 pairs, in place)
//   y    = flash-attention   (Q/K tiles via TDM, online softmax)
//   out  = y @ Wout + b      (same GEMM, fp32 out)
// ===========================================================================

typedef __attribute__((ext_vector_type(16))) __bf16 v16bf;
typedef __attribute__((ext_vector_type(8)))  float  v8f;
typedef __attribute__((ext_vector_type(4)))  unsigned int u32x4;
typedef __attribute__((ext_vector_type(4)))  int    i32x4;
typedef __attribute__((ext_vector_type(8)))  int    i32x8;

union Frag { v16bf v; unsigned int u[8]; };

// Pack two fp32 into one dword of bf16 (round-to-nearest-even).
__device__ __forceinline__ unsigned int bf16pair(float lo, float hi) {
  unsigned int a = __float_as_uint(lo);
  unsigned int b = __float_as_uint(hi);
  unsigned int ra = (a + 0x7FFFu + ((a >> 16) & 1u)) >> 16;
  unsigned int rb = (b + 0x7FFFu + ((b >> 16) & 1u));
  return (ra & 0xFFFFu) | (rb & 0xFFFF0000u);
}

// K-pair dword index held by operand VGPR v for a lane in half {0,1}
// (16-bit A/B operand layout, CDNA5 ISA 7.12.2).
__device__ __forceinline__ int kphalf(int v, int half) {
  return ((v & 4) << 1) + half * 4 + (v & 3);
}

__device__ __forceinline__ v8f wmma_bf16(v16bf a, v16bf b, v8f c) {
  return __builtin_amdgcn_wmma_f32_16x16x32_bf16(false, a, false, b, (short)0, c,
                                                 false, false);
}

// ---------------------------------------------------------------------------
// Tensor Data Mover: async 2D dword-tile load global -> LDS (TENSORcnt).
// D# per CDNA5 ISA. 2D: groups 2/3 zero, tile_dim2=0.
// ---------------------------------------------------------------------------
#if __has_builtin(__builtin_amdgcn_tensor_load_to_lds) && \
    __has_builtin(__builtin_amdgcn_s_wait_tensorcnt)
#define HAVE_TDM 1
#else
#define HAVE_TDM 0
#endif

#if HAVE_TDM
__device__ __forceinline__ void tdm_load_2d_dw(unsigned lds_off_bytes,
                                               unsigned long long gaddr,
                                               unsigned tile_w_dw,
                                               unsigned tile_h,
                                               unsigned long long row_stride_dw) {
  u32x4 g0 = {0u, 0u, 0u, 0u};
  g0.x = 1u;                                          // count=1, user descriptor
  g0.y = lds_off_bytes;                               // LDS byte offset
  g0.z = (unsigned)(gaddr & 0xFFFFFFFFull);           // global_addr[31:0]
  g0.w = (unsigned)((gaddr >> 32) & 0x01FFFFFFull)    // global_addr[56:32]
         | (2u << 30);                                // type=2 ("image")
  i32x8 g1 = {0, 0, 0, 0, 0, 0, 0, 0};
  g1[0] = (int)(2u << 16);                            // data_size = 4 bytes
  g1[1] = (int)((tile_w_dw & 0xFFFFu) << 16);         // tensor_dim0[15:0]
  g1[2] = (int)((tile_w_dw >> 16) | ((tile_h & 0xFFFFu) << 16)); // d0 hi|d1 lo
  g1[3] = (int)((tile_h >> 16) | (tile_w_dw << 16));  // d1 hi | tile_dim0
  g1[4] = (int)(tile_h & 0xFFFFu);                    // tile_dim1 (tile_dim2=0)
  g1[5] = (int)(row_stride_dw & 0xFFFFFFFFull);       // tensor_dim0_stride lo
  g1[6] = (int)(row_stride_dw >> 32);                 // stride hi (dim1_stride=0)
  i32x4 z4 = {0, 0, 0, 0};
#if defined(__clang_major__) && __clang_major__ >= 23
  i32x8 z8 = {0, 0, 0, 0, 0, 0, 0, 0};
  __builtin_amdgcn_tensor_load_to_lds(g0, g1, z4, z4, z8, 0);
#else
  __builtin_amdgcn_tensor_load_to_lds(g0, g1, z4, z4, 0);
#endif
}
__device__ __forceinline__ void tdm_wait0() {
  __builtin_amdgcn_s_wait_tensorcnt(0);
}
// Low 32 bits of a flat LDS pointer are the LDS byte offset (ISA §10.2).
__device__ __forceinline__ unsigned lds_off(const void* p) {
  return (unsigned)(unsigned long long)(uintptr_t)p;
}
#endif

// ---------------------------------------------------------------------------
// One-time packing kernels (amortized conversion -> zero VALU in GEMM loops).
// ---------------------------------------------------------------------------
// dst[i] = pack(src[2i], src[2i+1])  (row-contiguous K pairs, e.g. x, A-side)
__global__ __launch_bounds__(256) void pack_rows_kernel(
    const float* __restrict__ src, unsigned* __restrict__ dst, long n) {
  long i = (long)blockIdx.x * 256 + threadIdx.x;
  if (i < n) dst[i] = bf16pair(src[2 * i], src[2 * i + 1]);
}
// Wp[kp][n] = pack(W[2kp][n], W[2kp+1][n])  (K pairs gathered across rows, B-side)
__global__ __launch_bounds__(256) void pack_cols_kernel(
    const float* __restrict__ W, unsigned* __restrict__ Wp, int N, long n) {
  long i = (long)blockIdx.x * 256 + threadIdx.x;
  if (i < n) {
    long kp = i / N;
    long col = i - kp * N;
    Wp[i] = bf16pair(W[(2 * kp) * (long)N + col], W[(2 * kp + 1) * (long)N + col]);
  }
}

// ---------------------------------------------------------------------------
// GEMM: C[M,N] = A[M,K] @ B[K,N] + bias[N].
// A: bf16 K-pair dwords [M][K/2]; B: bf16 K-pair dwords [K/2][N].
// 256 threads = 8 wave32; tile 256(M) x 128(N); K-step 32 (16 pair-dwords).
// TDM double-buffer: wave0 prefetches step i+1's A/B tiles while all waves
// run 16 WMMA on step i. One barrier per K-step.
// ---------------------------------------------------------------------------
template <bool OUT_BF16>
__global__ __launch_bounds__(256) void gemm_bias_kernel(
    const unsigned* __restrict__ A, const unsigned* __restrict__ B,
    const float* __restrict__ bias, void* __restrict__ Cv,
    int M, int N, int K) {
  __shared__ unsigned As[2][256 * 16];  // [buf][row][16 k-pairs]
  __shared__ unsigned Bs[2][16 * 128];  // [buf][k-pair][128 cols]

  const int t = threadIdx.x;
  const int lane = t & 31, wave = t >> 5;
  const int half = lane >> 4, l16 = lane & 15;
  const int m0 = blockIdx.y * 256;
  const int n0 = blockIdx.x * 128;
  const int Kd = K >> 1;  // dwords per A row
  const int nk = K / 32;

  v8f acc[2][8];
#pragma unroll
  for (int i = 0; i < 2; ++i)
#pragma unroll
    for (int j = 0; j < 8; ++j) acc[i][j] = (v8f){};

#if HAVE_TDM
  if (wave == 0) {
    tdm_load_2d_dw(lds_off(&As[0][0]),
                   (unsigned long long)(uintptr_t)(A + (size_t)m0 * Kd),
                   16, 256, (unsigned long long)Kd);
    tdm_load_2d_dw(lds_off(&Bs[0][0]),
                   (unsigned long long)(uintptr_t)(B + n0),
                   128, 16, (unsigned long long)N);
  }
#endif

  for (int i = 0; i < nk; ++i) {
#if HAVE_TDM
    const int buf = i & 1;
    if (wave == 0) tdm_wait0();       // tiles for step i resident in LDS
    __syncthreads();                  // publish; also: buf^1 fully consumed
    if (wave == 0 && i + 1 < nk) {    // prefetch step i+1 while we compute
      int kp0 = (i + 1) * 16;
      tdm_load_2d_dw(lds_off(&As[buf ^ 1][0]),
                     (unsigned long long)(uintptr_t)(A + (size_t)m0 * Kd + kp0),
                     16, 256, (unsigned long long)Kd);
      tdm_load_2d_dw(lds_off(&Bs[buf ^ 1][0]),
                     (unsigned long long)(uintptr_t)(B + (size_t)kp0 * N + n0),
                     128, 16, (unsigned long long)N);
    }
#else
    const int buf = 0;
    __syncthreads();
    {
      int kp0 = i * 16;
#pragma unroll
      for (int j = 0; j < 16; ++j) {
        int idx = t + j * 256;
        int row = idx >> 4, kp = idx & 15;
        As[0][idx] = A[(size_t)(m0 + row) * Kd + kp0 + kp];
      }
#pragma unroll
      for (int j = 0; j < 8; ++j) {
        int idx = t + j * 256;
        int kp = idx >> 7, n = idx & 127;
        Bs[0][idx] = B[(size_t)(kp0 + kp) * N + n0 + n];
      }
    }
    __syncthreads();
#endif

    Frag af[2];
#pragma unroll
    for (int mt = 0; mt < 2; ++mt) {
      int m = wave * 32 + mt * 16 + l16;
#pragma unroll
      for (int v = 0; v < 8; ++v) af[mt].u[v] = As[buf][m * 16 + kphalf(v, half)];
    }
#pragma unroll
    for (int nt = 0; nt < 8; ++nt) {
      Frag bfr;
#pragma unroll
      for (int v = 0; v < 8; ++v)
        bfr.u[v] = Bs[buf][kphalf(v, half) * 128 + nt * 16 + l16];
#pragma unroll
      for (int mt = 0; mt < 2; ++mt)
        acc[mt][nt] = wmma_bf16(af[mt].v, bfr.v, acc[mt][nt]);
    }
  }

  // Epilogue. C layout: VGPR r -> row r+8*half, col = lane&15.
#pragma unroll
  for (int nt = 0; nt < 8; ++nt) {
    int col = n0 + nt * 16 + l16;
    float bv = bias[col];
#pragma unroll
    for (int mt = 0; mt < 2; ++mt)
#pragma unroll
      for (int r = 0; r < 8; ++r) {
        int row = m0 + wave * 32 + mt * 16 + r + 8 * half;
        float cv = acc[mt][nt][r] + bv;
        if (OUT_BF16) {
          float chi = __shfl_xor(cv, 1, 32);  // neighbor col (same half)
          if ((lane & 1) == 0)
            ((unsigned*)Cv)[(size_t)row * (N >> 1) + (col >> 1)] =
                bf16pair(cv, chi);
        } else {
          ((float*)Cv)[(size_t)row * N + col] = cv;
        }
      }
  }
}

// ---------------------------------------------------------------------------
// RoPE in place on bf16 qkv; each dword is exactly one rotation pair.
// qkv row layout (dwords): [q 0..1023 | k 1024..2047 | v 2048..3071].
// ---------------------------------------------------------------------------
__global__ __launch_bounds__(256) void rope_kernel(
    unsigned* __restrict__ qkv_dw, const float* __restrict__ fcos,
    const float* __restrict__ fsin, int S) {
  int bs = blockIdx.x;
  int ss = bs % S;
  unsigned* base = qkv_dw + (size_t)bs * 3072;
#pragma unroll
  for (int i = 0; i < 4; ++i) {
    int idx = threadIdx.x + i * 256;  // head*64 + pair
    int h = idx >> 6, p = idx & 63;
    float c = fcos[ss * 64 + p];
    float sn = fsin[ss * 64 + p];
    unsigned* q = base + h * 64 + p;
    unsigned u = *q;
    float xr = __uint_as_float(u << 16);
    float xi = __uint_as_float(u & 0xFFFF0000u);
    *q = bf16pair(xr * c - xi * sn, xr * sn + xi * c);
    unsigned* k = base + 1024 + h * 64 + p;
    u = *k;
    xr = __uint_as_float(u << 16);
    xi = __uint_as_float(u & 0xFFFF0000u);
    *k = bf16pair(xr * c - xi * sn, xr * sn + xi * c);
  }
}

// ---------------------------------------------------------------------------
// Flash attention (causal) per (b, h, 128-query tile). 8 waves x 16 q-rows.
// Q tile staged once by TDM; per 32-key step: TDM K tile + manual V pairing,
// 8 WMMA scores, shfl_xor online softmax, P->bf16 LDS, 8 WMMA for Y += P V.
// Emits y as packed bf16 pairs (feeds the output-projection GEMM).
// ---------------------------------------------------------------------------
__global__ __launch_bounds__(256) void flash_attn_kernel(
    const unsigned* __restrict__ qkv_dw, unsigned* __restrict__ Ydw, int S) {
  __shared__ unsigned Qs[128 * 64];    // 128 q-rows x 64 hd-pair dwords
  __shared__ unsigned Ks[32 * 64];     // 32 keys x 64 hd-pair dwords
  __shared__ unsigned Vs[16 * 128];    // 16 key-pair dwords x 128 hd cols
  __shared__ unsigned Ps[8][16 * 16];  // per-wave 16 rows x 16 key-pairs

  const int t = threadIdx.x;
  const int lane = t & 31, wave = t >> 5;
  const int half = lane >> 4, l16 = lane & 15;
  const int q0 = blockIdx.x * 128;
  const int h = blockIdx.y;
  const int bb = blockIdx.z;

  const size_t row0 = (size_t)bb * S;  // qkv row index base
  const unsigned* qrow = qkv_dw + row0 * 3072 + h * 64;         // q section
  const unsigned* krow = qkv_dw + row0 * 3072 + 1024 + h * 64;  // k section
  const unsigned short* v16 = (const unsigned short*)qkv_dw;    // element view

  // ---- stage Q tile (128 rows x 64 dwords, row stride 3072 dwords) ----
#if HAVE_TDM
  if (wave == 0) {
    tdm_load_2d_dw(lds_off(&Qs[0]),
                   (unsigned long long)(uintptr_t)(qrow + (size_t)q0 * 3072),
                   64, 128, 3072ull);
    tdm_wait0();
  }
  __syncthreads();
#else
#pragma unroll
  for (int i = 0; i < 32; ++i) {
    int idx = t + i * 256;
    int m = idx >> 6, p = idx & 63;
    Qs[idx] = qrow[(size_t)(q0 + m) * 3072 + p];
  }
  __syncthreads();
#endif

  Frag qf[4];
#pragma unroll
  for (int c = 0; c < 4; ++c) {
    int m = wave * 16 + l16;
#pragma unroll
    for (int v = 0; v < 8; ++v)
      qf[c].u[v] = Qs[m * 64 + c * 16 + kphalf(v, half)];
  }

  v8f yacc[8];
#pragma unroll
  for (int i = 0; i < 8; ++i) yacc[i] = (v8f){};
  float rowm[8], rowl[8];
#pragma unroll
  for (int r = 0; r < 8; ++r) { rowm[r] = -1e38f; rowl[r] = 0.0f; }

  const float sc = 0.12751745f;   // log2(e) / sqrt(128)
  const int nkt = (q0 >> 5) + 4;  // causal: keys [0, q0+128)

  for (int kt = 0; kt < nkt; ++kt) {
    const int kb = kt * 32;
    __syncthreads();  // previous K/V/P tiles fully consumed
#if HAVE_TDM
    if (wave == 0)
      tdm_load_2d_dw(lds_off(&Ks[0]),
                     (unsigned long long)(uintptr_t)(krow + (size_t)kb * 3072),
                     64, 32, 3072ull);
#else
#pragma unroll
    for (int i = 0; i < 8; ++i) {
      int idx = t + i * 256;
      int kk = idx >> 6, p = idx & 63;
      Ks[idx] = krow[(size_t)(kb + kk) * 3072 + p];
    }
#endif
    // V: pair along KEY dim -> Vs[kp][n] = {V[2kp][n], V[2kp+1][n]} (bf16).
#pragma unroll
    for (int i = 0; i < 8; ++i) {
      int idx = t + i * 256;
      int kp = idx >> 7, n = idx & 127;
      size_t e = (row0 + kb + kp * 2) * 6144 + 4096 + h * 128 + n;
      unsigned lo = v16[e], hi = v16[e + 6144];
      Vs[idx] = lo | (hi << 16);
    }
#if HAVE_TDM
    if (wave == 0) tdm_wait0();
#endif
    __syncthreads();

    // ---- scores: two 16(q) x 16(key) tiles, contraction over hd=128 ----
    float s[2][8];
#pragma unroll
    for (int nt = 0; nt < 2; ++nt) {
      v8f sacc = (v8f){};
#pragma unroll
      for (int c = 0; c < 4; ++c) {
        Frag kf;
#pragma unroll
        for (int v = 0; v < 8; ++v)
          kf.u[v] = Ks[(nt * 16 + l16) * 64 + c * 16 + kphalf(v, half)];
        sacc = wmma_bf16(qf[c].v, kf.v, sacc);
      }
      int ng = kb + nt * 16 + l16;
#pragma unroll
      for (int r = 0; r < 8; ++r) {
        int mg = q0 + wave * 16 + r + 8 * half;
        s[nt][r] = (ng <= mg) ? sacc[r] * sc : -1e38f;  // causal mask + scale
      }
    }

    // ---- online softmax; xor masks 1,2,4,8 reduce within each 16-lane half
    float p0[8], p1[8];
#pragma unroll
    for (int r = 0; r < 8; ++r) {
      float mx = fmaxf(s[0][r], s[1][r]);
      mx = fmaxf(mx, __shfl_xor(mx, 1, 32));
      mx = fmaxf(mx, __shfl_xor(mx, 2, 32));
      mx = fmaxf(mx, __shfl_xor(mx, 4, 32));
      mx = fmaxf(mx, __shfl_xor(mx, 8, 32));
      float mnew = fmaxf(rowm[r], mx);
      float scale = __builtin_amdgcn_exp2f(rowm[r] - mnew);
      p0[r] = __builtin_amdgcn_exp2f(s[0][r] - mnew);
      p1[r] = __builtin_amdgcn_exp2f(s[1][r] - mnew);
      float sum = p0[r] + p1[r];
      sum += __shfl_xor(sum, 1, 32);
      sum += __shfl_xor(sum, 2, 32);
      sum += __shfl_xor(sum, 4, 32);
      sum += __shfl_xor(sum, 8, 32);
      rowl[r] = rowl[r] * scale + sum;
      rowm[r] = mnew;
#pragma unroll
      for (int nt2 = 0; nt2 < 8; ++nt2) yacc[nt2][r] *= scale;
    }

    // ---- repack P (C layout) into A-operand bf16 pairs in LDS ----
#pragma unroll
    for (int r = 0; r < 8; ++r) {
      float h0 = __shfl_xor(p0[r], 1, 32);
      float h1 = __shfl_xor(p1[r], 1, 32);
      if ((lane & 1) == 0) {
        int m = r + 8 * half;
        Ps[wave][m * 16 + (l16 >> 1)] = bf16pair(p0[r], h0);
        Ps[wave][m * 16 + 8 + (l16 >> 1)] = bf16pair(p1[r], h1);
      }
    }
    __syncthreads();

    // ---- Y += P[16x32] @ V[32x128]: 8 WMMA ----
    Frag pf;
#pragma unroll
    for (int v = 0; v < 8; ++v)
      pf.u[v] = Ps[wave][l16 * 16 + kphalf(v, half)];
#pragma unroll
    for (int nt = 0; nt < 8; ++nt) {
      Frag vf;
#pragma unroll
      for (int v = 0; v < 8; ++v)
        vf.u[v] = Vs[kphalf(v, half) * 128 + nt * 16 + l16];
      yacc[nt] = wmma_bf16(pf.v, vf.v, yacc[nt]);
    }
  }

  // ---- finalize: /= rowsum, emit bf16 pairs into y [8192][1024] dwords ----
#pragma unroll
  for (int nt = 0; nt < 8; ++nt)
#pragma unroll
    for (int r = 0; r < 8; ++r) {
      int mg = q0 + wave * 16 + r + 8 * half;
      float val = yacc[nt][r] / rowl[r];
      float vhi = __shfl_xor(val, 1, 32);
      if ((lane & 1) == 0)
        Ydw[(row0 + mg) * 1024 + h * 64 + nt * 8 + (l16 >> 1)] =
            bf16pair(val, vhi);
    }
}

// ---------------------------------------------------------------------------
// Workspace (bytes): qkv bf16 96MB | y bf16 32MB | x bf16 32MB |
//                    WqkvP 24MB | WoutP 8MB   = 192MB total.
// ---------------------------------------------------------------------------
extern "C" void kernel_launch(void* const* d_in, const int* in_sizes, int n_in,
                              void* d_out, int out_size, void* d_ws, size_t ws_size,
                              hipStream_t stream) {
  const float* x    = (const float*)d_in[0];
  const float* fcos = (const float*)d_in[1];
  const float* fsin = (const float*)d_in[2];
  const float* Wqkv = (const float*)d_in[3];
  const float* bqkv = (const float*)d_in[4];
  const float* Wout = (const float*)d_in[5];
  const float* bout = (const float*)d_in[6];
  float* out = (float*)d_out;

  const int B = 4, S = 2048, D = 2048, D3 = 3 * D, H = 16;
  const int M = B * S;  // 8192

  unsigned* qkv_dw = (unsigned*)d_ws;                    // [M][3072] dwords
  unsigned* y_dw   = qkv_dw + (size_t)M * (D3 / 2);      // [M][1024]
  unsigned* xb     = y_dw + (size_t)M * (D / 2);         // [M][1024]
  unsigned* wqkvp  = xb + (size_t)M * (D / 2);           // [1024][6144]
  unsigned* woutp  = wqkvp + (size_t)(D / 2) * D3;       // [1024][2048]

  // 0) one-time bf16 packing (pairs along K for both GEMM operands)
  {
    long n = (long)M * (D / 2);
    pack_rows_kernel<<<dim3((unsigned)(n / 256)), 256, 0, stream>>>(x, xb, n);
  }
  {
    long n = (long)(D / 2) * D3;
    pack_cols_kernel<<<dim3((unsigned)(n / 256)), 256, 0, stream>>>(Wqkv, wqkvp, D3, n);
  }
  {
    long n = (long)(D / 2) * D;
    pack_cols_kernel<<<dim3((unsigned)(n / 256)), 256, 0, stream>>>(Wout, woutp, D, n);
  }
  // 1) qkv = x @ Wqkv + bqkv  (bf16 out)
  gemm_bias_kernel<true><<<dim3(D3 / 128, M / 256), 256, 0, stream>>>(
      xb, wqkvp, bqkv, qkv_dw, M, D3, D);
  // 2) RoPE on q, k (in place, packed pairs)
  rope_kernel<<<dim3(B * S), 256, 0, stream>>>(qkv_dw, fcos, fsin, S);
  // 3) y = causal_softmax(q k^T / sqrt(hd)) v
  flash_attn_kernel<<<dim3(S / 128, H, B), 256, 0, stream>>>(qkv_dw, y_dw, S);
  // 4) out = y @ Wout + bout  (fp32 out)
  gemm_bias_kernel<false><<<dim3(D / 128, M / 256), 256, 0, stream>>>(
      y_dw, woutp, bout, out, M, D, D);
}
